// H_attention_83408264888888
// MI455X (gfx1250) — matrix operand order repaired and checked by reading, not compiled, not run on hardware
//
#include <hip/hip_runtime.h>
#include <hip/hip_bf16.h>
#include <stdint.h>

// ---------------------------------------------------------------------------
// MI455X (gfx1250) implementation of the hierarchical-attention reference.
// All matmuls (linears + both attention GEMMs) run on v_wmma_f32_16x16x32_f16
// (f16 multiply, f32 accumulate). Attention is single-pass flash style so the
// 16x2048x2048 score tensor (268MB > 192MB L2) is never materialized.
// K/V tiles are staged into LDS with CDNA5 async global->LDS copies
// (global_load_async_to_lds_b128, ASYNCcnt) instead of going through VGPRs.
// wave32 everywhere: 128-thread blocks = 4 waves, one 16-row tile per wave.
// ---------------------------------------------------------------------------

typedef _Float16 half_t;
typedef half_t v16h __attribute__((ext_vector_type(16)));
typedef float  v8f  __attribute__((ext_vector_type(8)));

// 16-bit A-matrix 16x32 fragment: lane holds row M = lane&15; half j holds
// K = ((j<8)?0:16) + ((lane>=16)?8:0) + (j&7)   (ISA 7.12.2 table)
__device__ __forceinline__ int frag_ak(int lane, int j) {
  return ((j & 8) << 1) | ((lane & 16) >> 1) | (j & 7);
}
// 16-bit B-matrix 32x16 fragment: lane holds col N = lane&15; half j holds
// K = ((j<8)?0:16) + ((lane>=16)?8:0) + (j&7)   (same K order as A)
__device__ __forceinline__ int frag_bk(int lane, int j) {
  return ((j & 8) << 1) | ((lane & 16) >> 1) | (j & 7);
}

__device__ __forceinline__ v8f wmma_f16(v16h a, v16h b, v8f c) {
  return __builtin_amdgcn_wmma_f32_16x16x32_f16(
      /*neg_a=*/false, a, /*neg_b=*/false, b,
      /*c_mod=*/(short)0, c, /*reuse_a=*/false, /*reuse_b=*/false);
}

// CDNA5 async copy: 16B global -> LDS, tracked by ASYNCcnt (ISA 10.x / 15.18.3)
__device__ __forceinline__ void async_copy_b128(unsigned lds_off, const float* gaddr) {
  asm volatile("global_load_async_to_lds_b128 %0, %1, off"
               :: "v"(lds_off), "v"(gaddr) : "memory");
}
__device__ __forceinline__ void wait_asynccnt0() {
  asm volatile("s_wait_asynccnt 0x0" ::: "memory");
}
__device__ __forceinline__ unsigned lds_offset_of(const void* p) {
  // generic pointer to LDS: low 32 bits are the LDS byte offset
  return (unsigned)(uintptr_t)p;
}

// ---------------------------------------------------------------------------
// C[M,Nd] = A[M,K] @ W[K,Nd] + bias.  A may be a column-slice (lda), C may be
// written into a wider concat buffer (ldc).  grid = (Nd/16, M/64), block=128.
// K, Nd multiples of 32/16; M multiple of 64 (all shapes here qualify).
// ---------------------------------------------------------------------------
__global__ void gemm_bias_kernel(const float* __restrict__ A, int lda,
                                 const float* __restrict__ W,
                                 const float* __restrict__ bias,
                                 float* __restrict__ C, int ldc,
                                 int Nd, int K) {
  const int lane = threadIdx.x & 31;
  const int wave = threadIdx.x >> 5;
  const int row0 = (blockIdx.y * 4 + wave) * 16;
  const int col0 = blockIdx.x * 16;
  const int r  = lane & 15;
  const int hi = (lane >> 4) & 1;

  v8f acc = {};
  for (int k0 = 0; k0 < K; k0 += 32) {
    v16h a, b;
#pragma unroll
    for (int j = 0; j < 16; ++j) {
      a[j] = (half_t)A[(row0 + r) * lda + k0 + frag_ak(lane, j)];
      b[j] = (half_t)W[(k0 + frag_bk(lane, j)) * Nd + col0 + r];
    }
    acc = wmma_f16(a, b, acc);
  }
  const float bv = bias[col0 + r];
#pragma unroll
  for (int rr = 0; rr < 8; ++rr)
    C[(row0 + rr + hi * 8) * ldc + col0 + r] = acc[rr] + bv;
}

// ---------------------------------------------------------------------------
// Flash attention: O = softmax(Q K^T * scale) V, with NaN->0 like reference.
// One block (4 waves) = one batch x 64 query rows; each wave owns 16 rows.
// Key tiles of 32 staged into LDS (f32) via async b128 copies; converted to
// f16 at fragment-gather time.  q_bs==0 broadcasts Q across batch.
// grid = (16, nq/64), block = 128.
// ---------------------------------------------------------------------------
template <int DH>
__global__ void flash_attn_kernel(const float* __restrict__ Q, long q_bs,
                                  const float* __restrict__ Kp, long kv_bs,
                                  const float* __restrict__ Vp,
                                  float* __restrict__ O, long o_bs, int o_ld,
                                  int nk, float scale) {
  constexpr int KT  = 32;
  constexpr int NC  = DH / 32;  // K-chunks of the d-dimension per S WMMA
  constexpr int NT  = DH / 16;  // 16-wide output tiles
  constexpr int PAD = 8;        // 32B row pad (keeps 16B alignment for b128)
  __shared__ __align__(16) float Ks[KT][DH + PAD];
  __shared__ __align__(16) float Vs[KT][DH + PAD];
  __shared__ half_t Ps[4][16][32 + 8];

  const int tid  = threadIdx.x;
  const int lane = tid & 31;
  const int wave = tid >> 5;
  const int b    = blockIdx.x;
  const int r    = lane & 15;
  const int hi   = (lane >> 4) & 1;
  const int q0   = (blockIdx.y * 4 + wave) * 16;

  // Q fragments (A-layout), loaded once
  v16h qa[NC];
#pragma unroll
  for (int c = 0; c < NC; ++c)
#pragma unroll
    for (int j = 0; j < 16; ++j)
      qa[c][j] = (half_t)Q[q_bs * b + (q0 + r) * DH + c * 32 + frag_ak(lane, j)];

  v8f zero = {};
  v8f oacc[NT];
#pragma unroll
  for (int t = 0; t < NT; ++t) oacc[t] = zero;
  float m[8], l[8];
#pragma unroll
  for (int rr = 0; rr < 8; ++rr) { m[rr] = -3.0e38f; l[rr] = 0.0f; }

  for (int kt = 0; kt < nk; kt += KT) {
    __syncthreads();  // protect Ks/Vs against previous iteration's readers
    // async global->LDS staging: each thread copies 16B chunks of K and V
    constexpr int CH = KT * DH / 4;  // # of 16B chunks per matrix
#pragma unroll
    for (int c0 = 0; c0 < CH; c0 += 128) {
      int c    = c0 + tid;
      int row  = c / (DH / 4);
      int col4 = (c % (DH / 4)) * 4;
      long g   = kv_bs * b + (long)(kt + row) * DH + col4;
      async_copy_b128(lds_offset_of(&Ks[row][col4]), Kp + g);
      async_copy_b128(lds_offset_of(&Vs[row][col4]), Vp + g);
    }
    wait_asynccnt0();
    __syncthreads();

    // S tile: 16 queries x 32 keys as two 16x16 C-tiles
    v8f s0 = zero, s1 = zero;
#pragma unroll
    for (int c = 0; c < NC; ++c) {
      v16h b0, b1;
#pragma unroll
      for (int j = 0; j < 16; ++j) {
        int d = c * 32 + frag_bk(lane, j);
        b0[j] = (half_t)Ks[r][d];        // key cols 0-15
        b1[j] = (half_t)Ks[r + 16][d];   // key cols 16-31
      }
      s0 = wmma_f16(qa[c], b0, s0);
      s1 = wmma_f16(qa[c], b1, s1);
    }

    // online softmax; each row's 32 scores live in 16 lanes (2 regs each)
#pragma unroll
    for (int rr = 0; rr < 8; ++rr) {
      float v0 = s0[rr] * scale, v1 = s1[rr] * scale;
      float mx = fmaxf(v0, v1);
#pragma unroll
      for (int off = 1; off < 16; off <<= 1)
        mx = fmaxf(mx, __shfl_xor(mx, off, 32));
      float nm = fmaxf(m[rr], mx);
      float p0 = __expf(v0 - nm), p1 = __expf(v1 - nm);
      float ps = p0 + p1;
#pragma unroll
      for (int off = 1; off < 16; off <<= 1)
        ps += __shfl_xor(ps, off, 32);
      float corr = __expf(m[rr] - nm);
      l[rr] = l[rr] * corr + ps;
      m[rr] = nm;
#pragma unroll
      for (int t = 0; t < NT; ++t) oacc[t][rr] *= corr;
      s0[rr] = p0;
      s1[rr] = p1;
    }

    // re-layout P (C-layout) -> A-fragment via LDS
#pragma unroll
    for (int rr = 0; rr < 8; ++rr) {
      Ps[wave][rr + hi * 8][r]      = (half_t)s0[rr];
      Ps[wave][rr + hi * 8][r + 16] = (half_t)s1[rr];
    }
    __syncthreads();
    v16h pa;
#pragma unroll
    for (int j = 0; j < 16; ++j)
      pa[j] = Ps[wave][r][frag_ak(lane, j)];

#pragma unroll
    for (int t = 0; t < NT; ++t) {
      v16h bv;
#pragma unroll
      for (int j = 0; j < 16; ++j)
        bv[j] = (half_t)Vs[frag_bk(lane, j)][t * 16 + r];
      oacc[t] = wmma_f16(pa, bv, oacc[t]);
    }
  }

#pragma unroll
  for (int rr = 0; rr < 8; ++rr) {
    float inv = (l[rr] > 0.0f) ? 1.0f / l[rr] : 0.0f;
#pragma unroll
    for (int t = 0; t < NT; ++t) {
      float v = oacc[t][rr] * inv;
      if (v != v) v = 0.0f;  // jnp.where(isnan(o), 0, o)
      O[o_bs * b + (long)(q0 + rr + hi * 8) * o_ld + t * 16 + r] = v;
    }
  }
}

// ---------------------------------------------------------------------------
// Host orchestration. Input order (setup_inputs dict order):
//  0 data | lv0: 1 fc1_w 2 fc1_b 3 fc2_w 4 fc2_b 5 q_w 6 q_b 7 k_w 8 k_b
//  9 v_w 10 v_b | lv1: 11 fc1_w 12 fc1_b 13 fc2_w 14 fc2_b 15 emb 16 dq_w
//  17 dq_b 18 dk_w 19 dk_b 20 dv_w 21 dv_b 22 mq_w 23 mq_b 24 mk_w 25 mk_b
//  26 mv_w 27 mv_b 28 uq_w 29 uq_b 30 uk_w 31 uk_b 32 uv_w 33 uv_b |
//  lv2: 34..56 same layout as lv1.
// ---------------------------------------------------------------------------
extern "C" void kernel_launch(void* const* d_in, const int* in_sizes, int n_in,
                              void* d_out, int out_size, void* d_ws, size_t ws_size,
                              hipStream_t stream) {
  (void)in_sizes; (void)n_in; (void)out_size; (void)ws_size;
  auto F = [&](int i) { return (const float*)d_in[i]; };
  const int BN = 16 * 2048;

  float* p = (float*)d_ws;
  auto alloc = [&](long n) { float* r = p; p += n; return r; };
  float* bufA  = alloc((long)BN * 160);  // h0, then [a0 | out1]
  float* bufQ  = alloc((long)BN * 64);   // q0 -> k1 -> Ecat lo
  float* bufK  = alloc((long)BN * 64);   // k0 -> v1 -> Ecat hi
  float* bufV  = alloc((long)BN * 64);   // v0 -> q3 -> a2
  float* bufH1 = alloc((long)BN * 96);   // h1 -> {k1',v1',q3'}
  float* bufH2 = alloc((long)BN * 32);   // h2
  float* q1   = alloc(256L * 64);
  float* x1a  = alloc(4096L * 64);
  float* q2   = alloc(4096L * 64);
  float* k2   = alloc(4096L * 64);
  float* v2   = alloc(4096L * 64);
  float* x2a  = alloc(4096L * 64);
  float* k3   = alloc(4096L * 64);
  float* v3   = alloc(4096L * 64);
  float* q1p  = alloc(64L * 32);
  float* x1ap = alloc(1024L * 32);
  float* q2p  = alloc(1024L * 32);
  float* k2p  = alloc(1024L * 32);
  float* v2p  = alloc(1024L * 32);
  float* x2ap = alloc(1024L * 32);
  float* k3p  = alloc(1024L * 32);
  float* v3p  = alloc(1024L * 32);
  float* G0 = bufH1;                      // k1'  (h1 dead by then)
  float* G1 = bufH1 + (long)BN * 32;      // v1'
  float* G2 = bufH1 + (long)BN * 64;      // q3'
  float* Ecat = bufQ;                     // BN x 96 [a1 | out2], spans bufQ+bufK
  float* a2 = bufV;                       // BN x 32

  auto gemm = [&](const float* A_, int lda, const float* W_, const float* b_,
                  float* C_, int ldc, int M, int Nd, int K) {
    dim3 g(Nd / 16, M / 64), blk(128, 1, 1);
    gemm_bias_kernel<<<g, blk, 0, stream>>>(A_, lda, W_, b_, C_, ldc, Nd, K);
  };
  const float s64 = 0.125f;                 // 1/sqrt(64)
  const float s32 = 0.17677669529663687f;   // 1/sqrt(32)
  auto attn64 = [&](const float* Qp, long qbs, const float* Kp, long kvbs,
                    const float* Vp, float* O, long obs, int old_, int nq, int nk) {
    dim3 g(16, nq / 64), blk(128, 1, 1);
    flash_attn_kernel<64><<<g, blk, 0, stream>>>(Qp, qbs, Kp, kvbs, Vp, O, obs, old_, nk, s64);
  };
  auto attn32 = [&](const float* Qp, long qbs, const float* Kp, long kvbs,
                    const float* Vp, float* O, long obs, int old_, int nq, int nk) {
    dim3 g(16, nq / 64), blk(128, 1, 1);
    flash_attn_kernel<32><<<g, blk, 0, stream>>>(Qp, qbs, Kp, kvbs, Vp, O, obs, old_, nk, s32);
  };

  // ---- level 0 ----
  gemm(F(0), 160, F(1), F(2), bufA, 160, BN, 160, 160);            // h0
  gemm(bufA, 160, F(5), F(6), bufQ, 64, BN, 64, 64);               // q0
  gemm(bufA, 160, F(7), F(8), bufK, 64, BN, 64, 64);               // k0
  gemm(bufA, 160, F(9), F(10), bufV, 64, BN, 64, 64);              // v0
  attn64(bufQ, 2048L * 64, bufK, 2048L * 64, bufV,
         bufA, 2048L * 160, 160, 2048, 2048);                      // a0 -> bufA[:, :64]
  gemm(bufA + 64, 160, F(11), F(12), bufH1, 96, BN, 96, 96);       // h1

  // ---- level 1 ----
  gemm(F(15), 64, F(16), F(17), q1, 64, 256, 64, 64);              // q1 = emb @ dq
  gemm(bufH1, 96, F(18), F(19), bufQ, 64, BN, 64, 64);             // k1
  gemm(bufH1, 96, F(20), F(21), bufK, 64, BN, 64, 64);             // v1
  attn64(q1, 0, bufQ, 2048L * 64, bufK, x1a, 256L * 64, 64, 256, 2048);
  gemm(x1a, 64, F(22), F(23), q2, 64, 4096, 64, 64);               // mq
  gemm(x1a, 64, F(24), F(25), k2, 64, 4096, 64, 64);               // mk
  gemm(x1a, 64, F(26), F(27), v2, 64, 4096, 64, 64);               // mv
  attn64(q2, 256L * 64, k2, 256L * 64, v2, x2a, 256L * 64, 64, 256, 256);
  gemm(x2a, 64, F(30), F(31), k3, 64, 4096, 64, 64);               // uk
  gemm(x2a, 64, F(32), F(33), v3, 64, 4096, 64, 64);               // uv
  gemm(bufH1, 96, F(28), F(29), bufV, 64, BN, 64, 64);             // q3 = uq(x1)
  gemm(bufH1 + 64, 96, F(34), F(35), bufH2, 32, BN, 32, 32);       // h2 (last h1 read)
  attn64(bufV, 2048L * 64, k3, 256L * 64, v3,
         Ecat, 2048L * 96, 96, 2048, 256);                         // a1 -> Ecat[:, :64]

  // ---- level 2 ----
  gemm(F(38), 64, F(39), F(40), q1p, 32, 64, 32, 64);              // q1' = emb2 @ dq2
  gemm(bufH2, 32, F(41), F(42), G0, 32, BN, 32, 32);               // k1'
  gemm(bufH2, 32, F(43), F(44), G1, 32, BN, 32, 32);               // v1'
  attn32(q1p, 0, G0, 2048L * 32, G1, x1ap, 64L * 32, 32, 64, 2048);
  gemm(x1ap, 32, F(45), F(46), q2p, 32, 1024, 32, 32);             // mq2
  gemm(x1ap, 32, F(47), F(48), k2p, 32, 1024, 32, 32);             // mk2
  gemm(x1ap, 32, F(49), F(50), v2p, 32, 1024, 32, 32);             // mv2
  attn32(q2p, 64L * 32, k2p, 64L * 32, v2p, x2ap, 64L * 32, 32, 64, 64);
  gemm(x2ap, 32, F(53), F(54), k3p, 32, 1024, 32, 32);             // uk2
  gemm(x2ap, 32, F(55), F(56), v3p, 32, 1024, 32, 32);             // uv2
  gemm(bufH2, 32, F(51), F(52), G2, 32, BN, 32, 32);               // q3' = uq2(h2)
  attn32(G2, 2048L * 32, k3p, 64L * 32, v3p, a2, 2048L * 32, 32, 2048, 64);

  // ---- merge back up ----
  gemm(a2, 32, F(36), F(37), Ecat + 64, 96, BN, 32, 32);           // out2 -> Ecat[:, 64:]
  gemm(Ecat, 96, F(13), F(14), bufA + 64, 160, BN, 96, 96);        // out1 -> bufA[:, 64:]
  gemm(bufA, 160, F(3), F(4), (float*)d_out, 160, BN, 160, 160);   // final fc2_0
}
